// CustomMultiheadAttention_13657996001911
// MI455X (gfx1250) — matrix operand order repaired and checked by reading, not compiled, not run on hardware
//
#include <hip/hip_runtime.h>

// ---------------------------------------------------------------------------
// MHA forward for MI455X (gfx1250, wave32, WMMA).
// All GEMMs + attention matmuls on v_wmma_f32_16x16x32_bf16 (fp32 accumulate).
// Tile staging: CDNA5 async global->LDS loads (ASYNCcnt), double-buffered so
// the next tile streams in while WMMA consumes the current one (in-order
// ASYNCcnt completion lets "s_wait_asynccnt N" wait for the older tile only).
// Weights pre-transposed and V stored [b,h,d,s] so every WMMA fragment read
// from LDS is a contiguous ds_load_b128.
// ---------------------------------------------------------------------------

#define EMBED   1024
#define NHEADS  16
#define HDIM    64
#define BATCH   2
#define SEQ     2048
#define MTOT    (BATCH * SEQ)   // 4096 rows for projection GEMMs

typedef __attribute__((ext_vector_type(16))) __bf16       v16bf;
typedef __attribute__((ext_vector_type(8)))  float        v8f;

union BF16Frag { v16bf v; unsigned short u[16]; };

__device__ __forceinline__ unsigned short f32_to_bf16(float f) {
  unsigned int u = __float_as_uint(f);
  u += 0x7FFFu + ((u >> 16) & 1u);      // round-to-nearest-even
  return (unsigned short)(u >> 16);
}

// CDNA5 async global->LDS load (tracked by ASYNCcnt). GVS addressing:
// lds[vdst] = mem[saddr + vaddr], 16 bytes per lane.
// LDS byte offset = low 32 bits of the generic pointer (LDS aperture rule).
__device__ __forceinline__ void async_ld_b128(void* lds_dst, const void* base,
                                              unsigned byte_off) {
  unsigned l = (unsigned)(size_t)lds_dst;
  asm volatile("global_load_async_to_lds_b128 %0, %1, %2"
               :: "v"(l), "v"(byte_off), "s"(base)
               : "memory");
}
__device__ __forceinline__ void wait_async0() {   // all async loads done
  asm volatile("s_wait_asynccnt 0x0" ::: "memory");
}
__device__ __forceinline__ void wait_async4() {   // allow 4 (next GEMM tile) in flight
  asm volatile("s_wait_asynccnt 0x4" ::: "memory");
}
__device__ __forceinline__ void wait_async2() {   // allow 2 (next attn tile) in flight
  asm volatile("s_wait_asynccnt 0x2" ::: "memory");
}

// ---------------------------------------------------------------------------
// fp32 -> bf16 elementwise convert (activations)
// ---------------------------------------------------------------------------
__global__ __launch_bounds__(256) void cvt_f32_to_bf16(
    const float* __restrict__ in, unsigned short* __restrict__ out, int n) {
  int i = blockIdx.x * 256 + threadIdx.x;
  if (i < n) out[i] = f32_to_bf16(in[i]);
}

// ---------------------------------------------------------------------------
// Tiled transpose + convert: W (K x N, f32, row-major) -> Wt (N x K, bf16).
// ---------------------------------------------------------------------------
__global__ __launch_bounds__(256) void transpose_cvt(
    const float* __restrict__ W, unsigned short* __restrict__ Wt, int K, int N) {
  __shared__ float tile[32][33];
  const int kx = blockIdx.y * 32, nx = blockIdx.x * 32;
  const int tx = threadIdx.x & 31, ty = threadIdx.x >> 5;   // 32 x 8
#pragma unroll
  for (int r = 0; r < 32; r += 8)
    tile[ty + r][tx] = W[(size_t)(kx + ty + r) * N + nx + tx];
  __syncthreads();
#pragma unroll
  for (int r = 0; r < 32; r += 8)
    Wt[(size_t)(nx + ty + r) * K + kx + tx] = f32_to_bf16(tile[tx][ty + r]);
}

// ---------------------------------------------------------------------------
// Tiled bf16 WMMA GEMM:  out = A(MxK) * W(KxN) + bias, W given transposed.
// Block tile 128x128, 8 waves in a 4(M) x 2(N) grid, wave tile 32x64
// (8 WMMA per K-step of 32). Double-buffered async->LDS staging.
// MODE 0: bf16 out, split-head [b,h,s,d]          (Q, K projections)
// MODE 1: f32  out, plain row-major [M,N]         (final projection)
// MODE 2: bf16 out, transposed-head [b,h,d,s]     (V projection)
// ---------------------------------------------------------------------------
template <int MODE>
__global__ __launch_bounds__(256) void gemm_bf16_wmma(
    const unsigned short* __restrict__ A,    // M x K, bf16 row-major
    const unsigned short* __restrict__ Wt,   // N x K, bf16 row-major (W^T)
    const float* __restrict__ bias,          // N
    void* __restrict__ outp,
    int M, int N, int K) {
  __shared__ __align__(16) unsigned short As[2][128][40];  // 128x32 + pad, x2
  __shared__ __align__(16) unsigned short Bt[2][128][40];

  const int t  = threadIdx.x;
  const int w  = t >> 5, lane = t & 31;
  const int lr = lane & 15, hi = lane >> 4;
  const int wm = w >> 1, wn = w & 1;                       // wave grid 4(M) x 2(N)
  const int m0 = blockIdx.y * 128, n0 = blockIdx.x * 128;

  v8f c[2][4] = {};

  const int rowL = t >> 1, segL = (t & 1) * 16;            // 16 halves/thread/tile

  auto issue_tile = [&](int k0, int buf) {
    async_ld_b128(&As[buf][rowL][segL],     A,  (unsigned)(((size_t)(m0 + rowL) * K + k0 + segL) * 2));
    async_ld_b128(&As[buf][rowL][segL + 8], A,  (unsigned)(((size_t)(m0 + rowL) * K + k0 + segL + 8) * 2));
    async_ld_b128(&Bt[buf][rowL][segL],     Wt, (unsigned)(((size_t)(n0 + rowL) * K + k0 + segL) * 2));
    async_ld_b128(&Bt[buf][rowL][segL + 8], Wt, (unsigned)(((size_t)(n0 + rowL) * K + k0 + segL + 8) * 2));
  };

  const int nIt = K / 32;
  issue_tile(0, 0);                                        // pipeline prologue

  for (int it = 0; it < nIt; ++it) {
    const int cur = it & 1;
    if (it + 1 < nIt) { issue_tile((it + 1) * 32, cur ^ 1); wait_async4(); }
    else              { wait_async0(); }
    __syncthreads();                                       // tile `it` visible to all

    BF16Frag a[2], b[4];
    // A fragment: lane = row; 16 consecutive K -> contiguous 2x b128
#pragma unroll
    for (int i = 0; i < 2; ++i) {
      const unsigned short* row = &As[cur][wm * 32 + i * 16 + lr][hi * 16];
#pragma unroll
      for (int e = 0; e < 8; ++e) { a[i].u[e] = row[e]; a[i].u[8 + e] = row[8 + e]; }
    }
    // B fragment: lane = output column; 16 consecutive K -> contiguous 2x b128
#pragma unroll
    for (int j = 0; j < 4; ++j) {
      const unsigned short* col = &Bt[cur][wn * 64 + j * 16 + lr][hi * 16];
#pragma unroll
      for (int e = 0; e < 16; ++e) b[j].u[e] = col[e];
    }
#pragma unroll
    for (int i = 0; i < 2; ++i)
#pragma unroll
      for (int j = 0; j < 4; ++j)
        c[i][j] = __builtin_amdgcn_wmma_f32_16x16x32_bf16(
            false, a[i].v, false, b[j].v, (short)0, c[i][j], false, false);

    __syncthreads();                                       // buffer reuse guard
  }

  // Epilogue. C layout: VGPR e -> row e + 8*hi, col = lane&15.
#pragma unroll
  for (int i = 0; i < 2; ++i)
#pragma unroll
    for (int j = 0; j < 4; ++j)
#pragma unroll
      for (int e = 0; e < 8; ++e) {
        const int gm = m0 + wm * 32 + i * 16 + e + 8 * hi;
        const int gn = n0 + wn * 64 + j * 16 + lr;
        const float val = c[i][j][e] + bias[gn];
        if (MODE == 0) {
          const int bb = gm / SEQ, ss = gm % SEQ;
          const int hh = gn >> 6, dd = gn & 63;
          ((unsigned short*)outp)[(((size_t)bb * NHEADS + hh) * SEQ + ss) * HDIM + dd] =
              f32_to_bf16(val);
        } else if (MODE == 2) {
          const int bb = gm / SEQ, ss = gm % SEQ;
          const int hh = gn >> 6, dd = gn & 63;
          ((unsigned short*)outp)[(((size_t)bb * NHEADS + hh) * HDIM + dd) * SEQ + ss] =
              f32_to_bf16(val);
        } else {
          ((float*)outp)[(size_t)gm * N + gn] = val;
        }
      }
}

// ---------------------------------------------------------------------------
// Flash-style attention. One block = one (b,h) + 128 q-rows (8 waves x 16).
// kv in blocks of 32; K tile [kv][d] and V tile [d][kv] staged with
// double-buffered async->LDS loads. Per kv block, per wave: 4 WMMA scores,
// online softmax (16-lane shfl_xor, wave32), P bounced through wave-private
// LDS into an A fragment, 4 WMMA P.V.
// ---------------------------------------------------------------------------
__global__ __launch_bounds__(256) void attn_fwd(
    const unsigned short* __restrict__ Qp,   // [b,h,s,d] bf16
    const unsigned short* __restrict__ Kp,   // [b,h,s,d] bf16
    const unsigned short* __restrict__ Vt,   // [b,h,d,s] bf16 (transposed)
    unsigned short* __restrict__ ctx) {      // [b,s,e] bf16
  __shared__ __align__(16) unsigned short Ks[2][32][72];   // [kv][d] 32x64 + pad
  __shared__ __align__(16) unsigned short Vs[2][64][40];   // [d][kv] 64x32 + pad
  __shared__ __align__(16) unsigned short Ps[8][16][32];   // per-wave P staging

  const int t = threadIdx.x, w = t >> 5, lane = t & 31;
  const int lr = lane & 15, hi = lane >> 4;

  const int nqblk = SEQ / 128;                     // 16
  const int bh = blockIdx.x / nqblk;
  const int qblk = blockIdx.x % nqblk;
  const unsigned short* Qh = Qp + (size_t)bh * SEQ * HDIM;
  const unsigned short* Kh = Kp + (size_t)bh * SEQ * HDIM;
  const unsigned short* Vh = Vt + (size_t)bh * HDIM * SEQ;
  const int q0 = qblk * 128 + w * 16;

  // Wave's Q tile (16 x 64) as two A fragments (K-chunks of 32).
  BF16Frag aq[2];
  {
    const unsigned short* qrow = Qh + (size_t)(q0 + lr) * HDIM;
#pragma unroll
    for (int cc = 0; cc < 2; ++cc) {
      const int base = cc * 32 + hi * 16;
#pragma unroll
      for (int e = 0; e < 8; ++e) {
        aq[cc].u[e] = qrow[base + e];
        aq[cc].u[8 + e] = qrow[base + 8 + e];
      }
    }
  }

  float mrow[8], lrow[8];
  v8f o[4] = {};
#pragma unroll
  for (int e = 0; e < 8; ++e) { mrow[e] = -3.0e38f; lrow[e] = 0.0f; }

  const float sc = 0.125f;                         // 1/sqrt(HDIM)
  const int krow = t >> 3, kseg = (t & 7) * 8;     // K tile: 32 rows x 64 d
  const int vrow = t >> 2, vseg = (t & 3) * 8;     // V tile: 64 d-rows x 32 kv

  auto issue_tile = [&](int kv0, int buf) {
    async_ld_b128(&Ks[buf][krow][kseg], Kh, (unsigned)(((size_t)(kv0 + krow) * HDIM + kseg) * 2));
    async_ld_b128(&Vs[buf][vrow][vseg], Vh, (unsigned)(((size_t)vrow * SEQ + kv0 + vseg) * 2));
  };

  const int nIt = SEQ / 32;                        // 64
  issue_tile(0, 0);                                // pipeline prologue

  for (int it = 0; it < nIt; ++it) {
    const int cur = it & 1;
    if (it + 1 < nIt) { issue_tile((it + 1) * 32, cur ^ 1); wait_async2(); }
    else              { wait_async0(); }
    __syncthreads();                               // kv tile `it` visible

    // ---- scores: S = Q . K^T (16 q-rows x 32 kv-cols) ----
    v8f sfr[2];
#pragma unroll
    for (int n = 0; n < 2; ++n) {
      v8f acc = {};
#pragma unroll
      for (int cc = 0; cc < 2; ++cc) {
        BF16Frag bk;   // B[kdim][kvcol] = K[kvcol][kdim] -> contiguous d reads
        const unsigned short* kr = &Ks[cur][n * 16 + lr][cc * 32 + hi * 16];
#pragma unroll
        for (int e = 0; e < 16; ++e) bk.u[e] = kr[e];
        acc = __builtin_amdgcn_wmma_f32_16x16x32_bf16(
            false, aq[cc].v, false, bk.v, (short)0, acc, false, false);
      }
#pragma unroll
      for (int e = 0; e < 8; ++e) acc[e] *= sc;
      sfr[n] = acc;
    }

    // ---- online softmax (row r lives on the 16 lanes of one half-wave) ----
    float corr[8];
#pragma unroll
    for (int e = 0; e < 8; ++e) {
      float v = fmaxf(sfr[0][e], sfr[1][e]);
      v = fmaxf(v, __shfl_xor(v, 1, 16));
      v = fmaxf(v, __shfl_xor(v, 2, 16));
      v = fmaxf(v, __shfl_xor(v, 4, 16));
      v = fmaxf(v, __shfl_xor(v, 8, 16));
      const float nm = fmaxf(mrow[e], v);
      corr[e] = __expf(mrow[e] - nm);
      mrow[e] = nm;
    }
#pragma unroll
    for (int n = 0; n < 2; ++n)
#pragma unroll
      for (int e = 0; e < 8; ++e) sfr[n][e] = __expf(sfr[n][e] - mrow[e]);
#pragma unroll
    for (int e = 0; e < 8; ++e) {
      float s = sfr[0][e] + sfr[1][e];
      s += __shfl_xor(s, 1, 16);
      s += __shfl_xor(s, 2, 16);
      s += __shfl_xor(s, 4, 16);
      s += __shfl_xor(s, 8, 16);
      lrow[e] = lrow[e] * corr[e] + s;
    }
#pragma unroll
    for (int d = 0; d < 4; ++d)
#pragma unroll
      for (int e = 0; e < 8; ++e) o[d][e] *= corr[e];

    // ---- P: C-layout -> A-fragment via wave-private LDS bounce ----
#pragma unroll
    for (int n = 0; n < 2; ++n)
#pragma unroll
      for (int e = 0; e < 8; ++e)
        Ps[w][e + 8 * hi][n * 16 + lr] = f32_to_bf16(sfr[n][e]);

    BF16Frag ap;
    {
      const unsigned short* prow = &Ps[w][lr][hi * 16];
#pragma unroll
      for (int e = 0; e < 8; ++e) { ap.u[e] = prow[e]; ap.u[8 + e] = prow[8 + e]; }
    }

    // ---- O += P . V  (V fragment: lane = d column; contiguous kv reads) ----
#pragma unroll
    for (int d = 0; d < 4; ++d) {
      BF16Frag bv;
      const unsigned short* vr = &Vs[cur][d * 16 + lr][hi * 16];
#pragma unroll
      for (int e = 0; e < 16; ++e) bv.u[e] = vr[e];
      o[d] = __builtin_amdgcn_wmma_f32_16x16x32_bf16(
          false, ap.v, false, bv.v, (short)0, o[d], false, false);
    }

    __syncthreads();                               // buffer reuse guard
  }

  // ---- epilogue: normalize, write ctx[b, s, h*64 + d] ----
  const int hh = bh % NHEADS, bb = bh / NHEADS;
#pragma unroll
  for (int e = 0; e < 8; ++e) {
    const float inv = 1.0f / lrow[e];
    const int srow = q0 + e + 8 * hi;
    unsigned short* orow = ctx + ((size_t)bb * SEQ + srow) * EMBED + hh * HDIM;
#pragma unroll
    for (int d = 0; d < 4; ++d) orow[d * 16 + lr] = f32_to_bf16(o[d][e] * inv);
  }
}

// ---------------------------------------------------------------------------
// Launcher
// ---------------------------------------------------------------------------
extern "C" void kernel_launch(void* const* d_in, const int* in_sizes, int n_in,
                              void* d_out, int out_size, void* d_ws, size_t ws_size,
                              hipStream_t stream) {
  (void)in_sizes; (void)n_in; (void)out_size; (void)ws_size;
  const float* query = (const float*)d_in[0];
  const float* key_t = (const float*)d_in[1];
  const float* value = (const float*)d_in[2];
  const float* Wq = (const float*)d_in[3];
  const float* bq = (const float*)d_in[4];
  const float* Wk = (const float*)d_in[5];
  const float* bk = (const float*)d_in[6];
  const float* Wv = (const float*)d_in[7];
  const float* bv = (const float*)d_in[8];
  const float* Wo = (const float*)d_in[9];
  const float* bo = (const float*)d_in[10];

  const size_t NE = (size_t)BATCH * SEQ * EMBED;  // 4,194,304 activations
  const size_t WE = (size_t)EMBED * EMBED;        // 1,048,576 weights

  unsigned short* ws  = (unsigned short*)d_ws;    // 64 MB total bf16 scratch
  unsigned short* qb  = ws;                       // [b,s,e]
  unsigned short* kb  = qb + NE;
  unsigned short* vb  = kb + NE;
  unsigned short* wqb = vb + NE;                  // W^T  (N x K)
  unsigned short* wkb = wqb + WE;
  unsigned short* wvb = wkb + WE;
  unsigned short* wob = wvb + WE;
  unsigned short* Qp  = wob + WE;                 // [b,h,s,d]
  unsigned short* Kp  = Qp + NE;                  // [b,h,s,d]
  unsigned short* Vp  = Kp + NE;                  // [b,h,d,s]
  unsigned short* ctx = Vp + NE;                  // [b,s,e]

  const dim3 blk(256);
  cvt_f32_to_bf16<<<(NE + 255) / 256, blk, 0, stream>>>(query, qb, (int)NE);
  cvt_f32_to_bf16<<<(NE + 255) / 256, blk, 0, stream>>>(key_t, kb, (int)NE);
  cvt_f32_to_bf16<<<(NE + 255) / 256, blk, 0, stream>>>(value, vb, (int)NE);

  const dim3 tgrid(EMBED / 32, EMBED / 32);
  transpose_cvt<<<tgrid, blk, 0, stream>>>(Wq, wqb, EMBED, EMBED);
  transpose_cvt<<<tgrid, blk, 0, stream>>>(Wk, wkb, EMBED, EMBED);
  transpose_cvt<<<tgrid, blk, 0, stream>>>(Wv, wvb, EMBED, EMBED);
  transpose_cvt<<<tgrid, blk, 0, stream>>>(Wo, wob, EMBED, EMBED);

  const dim3 ggrid(EMBED / 128, MTOT / 128);      // (8, 32)
  gemm_bf16_wmma<0><<<ggrid, blk, 0, stream>>>(qb, wqb, bq, Qp, MTOT, EMBED, EMBED);
  gemm_bf16_wmma<0><<<ggrid, blk, 0, stream>>>(kb, wkb, bk, Kp, MTOT, EMBED, EMBED);
  gemm_bf16_wmma<2><<<ggrid, blk, 0, stream>>>(vb, wvb, bv, Vp, MTOT, EMBED, EMBED);

  attn_fwd<<<BATCH * NHEADS * (SEQ / 128), blk, 0, stream>>>(Qp, Kp, Vp, ctx);

  gemm_bf16_wmma<1><<<ggrid, blk, 0, stream>>>(ctx, wob, bo, d_out, MTOT, EMBED, EMBED);
}
